// GatedMultiHeadAttention_17927193493826
// MI455X (gfx1250) — matrix operand order repaired and checked by reading, not compiled, not run on hardware
//
#include <hip/hip_runtime.h>

typedef __attribute__((ext_vector_type(16))) _Float16 v16h;
typedef __attribute__((ext_vector_type(8)))  _Float16 v8h;
typedef __attribute__((ext_vector_type(8)))  float    v8f;
typedef __attribute__((ext_vector_type(4)))  unsigned int v4u;
typedef __attribute__((ext_vector_type(8)))  int      v8i;
typedef __attribute__((ext_vector_type(4)))  int      v4i;

#define BB 4
#define SS 2048
#define DD 1024
#define HH 8
#define DH 128
#define MM 4
#define ATT_SCALE 0.08838834764831845f  // 128^-0.5

#if defined(__gfx1250__) && __has_builtin(__builtin_amdgcn_tensor_load_to_lds) && \
    __has_builtin(__builtin_amdgcn_s_wait_tensorcnt)
#define USE_TDM 1
#else
#define USE_TDM 0
#endif

__device__ __forceinline__ v8f wmma16(v16h a, v16h b, v8f c) {
  // D(16x16 f32) = A(16x32 f16) * B(32x16 f16) + C
  return __builtin_amdgcn_wmma_f32_16x16x32_f16(false, a, false, b, (short)0, c, false, false);
}

// A-fragment loader: a[0..7] from p0, a[8..15] from p1 (each 8 contiguous halfs)
__device__ __forceinline__ v16h load8x2(const _Float16* p0, const _Float16* p1) {
  v8h a = *(const v8h*)p0;
  v8h b = *(const v8h*)p1;
  v16h r;
#pragma unroll
  for (int i = 0; i < 8; ++i) { r[i] = a[i]; r[i + 8] = b[i]; }
  return r;
}

#if USE_TDM
// TDM: stage a 32(k) x 16(n) fp32 tile of W into LDS offset 0, padded to
// row stride 17 floats (pad_interval=16 DWORDs -> +1 DWORD per row).
__device__ __forceinline__ void tdm_stage_w(const float* W, int k0, int n0) {
  unsigned long long ga = (unsigned long long)W + (((unsigned long long)k0 * DD + n0) << 2);
  v4u g0;
  g0[0] = 1u;                                   // count=1, user descriptor
  g0[1] = 0u;                                   // lds_addr = 0 (tile at LDS base)
  g0[2] = (unsigned int)(ga & 0xffffffffu);     // global_addr[31:0]
  g0[3] = (unsigned int)((ga >> 32) & 0x01ffffffu) | (2u << 30);  // addr[56:32], type=2
  v8i g1;
  g1[0] = (2 << 16) | (1 << 20) | (3 << 22);    // data_size=4B, pad_enable, pad_interval=16DW
  g1[1] = (DD & 0xffff) << 16;                  // tensor_dim0 = 1024 (low 16)
  g1[2] = 0;                                    // dim0 hi = 0, tensor_dim1 low16 = 0
  g1[3] = 16 | (16 << 16);                      // tensor_dim1 = 1<<20 (hi16=16); tile_dim0=16
  g1[4] = 32;                                   // tile_dim1=32, tile_dim2=0
  g1[5] = DD;                                   // tensor_dim0_stride = 1024
  g1[6] = 0;
  g1[7] = 0;
  v4i gz = {0, 0, 0, 0};
#if __clang_major__ >= 23
  v8i gz8 = {0, 0, 0, 0, 0, 0, 0, 0};
  __builtin_amdgcn_tensor_load_to_lds(g0, g1, gz, gz, gz8, 0);
#else
  __builtin_amdgcn_tensor_load_to_lds(g0, g1, gz, gz, 0);
#endif
  __builtin_amdgcn_s_wait_tensorcnt(0);
}
#endif

// ---------------- kernel 1: xmean[b][d] = mean_s x[b][s][d] ----------------
__global__ void mean_k(const float* __restrict__ x, float* __restrict__ xmean) {
  int bx = blockIdx.x;                 // 16 blocks: b*4 + dchunk
  int b = bx >> 2;
  int d = ((bx & 3) << 8) + threadIdx.x;
  const float* p = x + (size_t)b * SS * DD + d;
  float s = 0.f;
  for (int t = 0; t < SS; ++t) s += p[(size_t)t * DD];
  xmean[b * DD + d] = s * (1.0f / SS);
}

// ---------------- kernel 2: gate = softmax(xmean @ Wg + bg) over H*M=32 ----
__global__ void gate_k(const float* __restrict__ xmean, const float* __restrict__ Wg,
                       const float* __restrict__ bg, float* __restrict__ gate) {
  int tid = threadIdx.x;               // 128 threads: one wave per batch
  int b = tid >> 5;
  int j = tid & 31;
  float acc = bg[j];
  for (int d = 0; d < DD; ++d) acc += xmean[b * DD + d] * Wg[d * 32 + j];
  float mx = acc;
#pragma unroll
  for (int m = 16; m >= 1; m >>= 1) mx = fmaxf(mx, __shfl_xor(mx, m, 32));
  float e = __expf(acc - mx);
  float s = e;
#pragma unroll
  for (int m = 16; m >= 1; m >>= 1) s += __shfl_xor(s, m, 32);
  gate[b * 32 + j] = e / s;
}

// ---------------- kernels 3/5: GEMM Y = A @ W + bias, W staged in LDS via TDM
// Block = 8 waves = 128(m) x 16(n) output slab; per k-step one 32x16 W tile
// is staged (padded to 17-float rows) and shared by all 8 waves.
// mode 0: Y f16 row-major; mode 1: per-head transposed Vt f16; mode 2: f32 out
template <typename TA>
__global__ void __launch_bounds__(256) proj_k(const TA* __restrict__ X,
                                              const float* __restrict__ W,
                                              const float* __restrict__ bias,
                                              _Float16* __restrict__ Yh,
                                              _Float16* __restrict__ Yt,
                                              float* __restrict__ Yf,
                                              int mode) {
  __shared__ float wtile[32 * 17];
  int tid  = threadIdx.x;
  int lane = tid & 31;
  int wave = tid >> 5;
  int nblk = blockIdx.x & 63;
  int mblk = blockIdx.x >> 6;
  int m0 = (mblk << 7) + (wave << 4);
  int n0 = nblk << 4;
  int grp = lane >> 4, l16 = lane & 15;
  int arow = m0 + l16;
  int bcol = n0 + l16;
  v8f acc = {};
  for (int k0 = 0; k0 < DD; k0 += 32) {
    __syncthreads();                       // previous tile fully consumed
#if USE_TDM
    if (tid == 0) tdm_stage_w(W, k0, n0);  // async DMA + s_wait_tensorcnt
#else
#pragma unroll
    for (int e = tid; e < 512; e += 256) {
      int kr = e >> 4, c = e & 15;
      wtile[kr * 17 + c] = W[(size_t)(k0 + kr) * DD + n0 + c];
    }
#endif
    __syncthreads();
    const TA* ap = X + (size_t)arow * DD + k0 + grp * 8;
    v16h va, vb;
#pragma unroll
    for (int i = 0; i < 8; ++i) {
      va[i]     = (_Float16)ap[i];
      va[i + 8] = (_Float16)ap[i + 16];
    }
#pragma unroll
    for (int i = 0; i < 16; ++i)
      vb[i] = (_Float16)wtile[((i & 7) + grp * 8 + (i >> 3) * 16) * 17 + l16];
    acc = wmma16(va, vb, acc);
  }
  float bn = bias[bcol];
#pragma unroll
  for (int r = 0; r < 8; ++r) {
    int rc = m0 + grp * 8 + r;
    float v = acc[r] + bn;
    if (mode == 0) {
      Yh[(size_t)rc * DD + bcol] = (_Float16)v;
    } else if (mode == 1) {
      int bb = rc >> 11;          // / S
      int s  = rc & (SS - 1);
      int h  = bcol >> 7;         // / DH
      int dd = bcol & (DH - 1);
      Yt[(size_t)(((bb << 3) + h) * DH + dd) * SS + s] = (_Float16)v;
    } else {
      Yf[(size_t)rc * DD + bcol] = v;
    }
  }
}

// ---------------- kernel 4: flash attention, 1 wave = 16 query rows of (b,h)
__global__ void __launch_bounds__(128) attn_k(const _Float16* __restrict__ Qh,
                                              const _Float16* __restrict__ Kh,
                                              const _Float16* __restrict__ Vt,
                                              const int* __restrict__ mod,
                                              const float* __restrict__ gate,
                                              _Float16* __restrict__ Oh) {
  int lane = threadIdx.x & 31;
  int flat = blockIdx.x * 4 + (threadIdx.x >> 5); // 4096 tasks
  int itile = flat & 127;
  int h = (flat >> 7) & 7;
  int b = flat >> 10;
  int i0 = itile << 4;
  int grp = lane >> 4, l16 = lane & 15;
  int i = i0 + l16;                     // this lane's query row (St column)
  int mod_i = mod[b * SS + i];
  float gi = 0.5f * gate[(b * HH + h) * MM + mod_i];

  // Q as B-fragments (4 k-chunks of 32 over DH=128): b[idx]=Q[i][d0+grp*8+(idx&7)+(idx>>3)*16]
  v16h vbq[4];
#pragma unroll
  for (int kk = 0; kk < 4; ++kk) {
    const _Float16* qp = Qh + (size_t)(b * SS + i) * DD + h * DH + kk * 32 + grp * 8;
    vbq[kk] = load8x2(qp, qp + 16);
  }

  v8f acc[8];
#pragma unroll
  for (int t = 0; t < 8; ++t) acc[t] = (v8f){};
  float mV = -1e30f, lV = 0.f;

  for (int j0 = 0; j0 < SS; j0 += 32) {
    // transposed score tiles St[j][i] = sum_d K[j][d] Q[i][d]
    v8f c0 = {}, c1 = {};
#pragma unroll
    for (int kk = 0; kk < 4; ++kk) {
      const _Float16* kp = Kh + (size_t)(b * SS + j0 + l16) * DD + h * DH + kk * 32 + grp * 8;
      c0 = wmma16(load8x2(kp, kp + 16), vbq[kk], c0);
    }
#pragma unroll
    for (int kk = 0; kk < 4; ++kk) {
      const _Float16* kp = Kh + (size_t)(b * SS + j0 + 16 + l16) * DD + h * DH + kk * 32 + grp * 8;
      c1 = wmma16(load8x2(kp, kp + 16), vbq[kk], c1);
    }
    // scale + modality-gate bias; St c[r] -> (j = j0 + grp*8 + r, col i)
    float p0[8], p1[8];
    float tmax = -1e30f;
#pragma unroll
    for (int r = 0; r < 8; ++r) {
      int j = j0 + grp * 8 + r;
      float s0 = c0[r] * ATT_SCALE + ((mod[b * SS + j] == mod_i) ? gi : 0.f);
      float s1 = c1[r] * ATT_SCALE + ((mod[b * SS + j + 16] == mod_i) ? gi : 0.f);
      p0[r] = s0; p1[r] = s1;
      tmax = fmaxf(tmax, fmaxf(s0, s1));
    }
    tmax = fmaxf(tmax, __shfl_xor(tmax, 16, 32));
    float mnew = fmaxf(mV, tmax);
    float alpha = __expf(mV - mnew);
    float lsum = 0.f;
#pragma unroll
    for (int r = 0; r < 8; ++r) {
      p0[r] = __expf(p0[r] - mnew);
      p1[r] = __expf(p1[r] - mnew);
      lsum += p0[r] + p1[r];
    }
    lsum += __shfl_xor(lsum, 16, 32);
    lV = lV * alpha + lsum;
    mV = mnew;
#pragma unroll
    for (int t = 0; t < 8; ++t)
#pragma unroll
      for (int r = 0; r < 8; ++r) acc[t][r] *= alpha;
    // pack P^T B-fragment: b[idx] = P[i][j0 + grp*16 + idx]
    v16h bp;
#pragma unroll
    for (int r = 0; r < 8; ++r) {
      float o0 = __shfl_xor(p0[r], 16, 32);
      float o1 = __shfl_xor(p1[r], 16, 32);
      bp[r]     = (_Float16)(grp ? o1 : p0[r]);
      bp[r + 8] = (_Float16)(grp ? p1[r] : o0);
    }
    // O^T += V^T * P^T  (8 d-tiles of 16)
#pragma unroll
    for (int t = 0; t < 8; ++t) {
      const _Float16* vp = Vt + (size_t)((b * HH + h) * DH + t * 16 + l16) * SS + j0 + grp * 16;
      acc[t] = wmma16(load8x2(vp, vp + 8), bp, acc[t]);
    }
  }

  float inv = 1.f / lV;
#pragma unroll
  for (int t = 0; t < 8; ++t)
#pragma unroll
    for (int r = 0; r < 8; ++r) {
      int d = t * 16 + grp * 8 + r;
      Oh[(size_t)(b * SS + i) * DD + h * DH + d] = (_Float16)(acc[t][r] * inv);
    }
}

extern "C" void kernel_launch(void* const* d_in, const int* in_sizes, int n_in,
                              void* d_out, int out_size, void* d_ws, size_t ws_size,
                              hipStream_t stream) {
  (void)in_sizes; (void)n_in; (void)out_size; (void)ws_size;
  const float* x   = (const float*)d_in[0];
  const int*   mid = (const int*)d_in[1];
  const float* Wq  = (const float*)d_in[2];
  const float* bq  = (const float*)d_in[3];
  const float* Wk  = (const float*)d_in[4];
  const float* bk  = (const float*)d_in[5];
  const float* Wv  = (const float*)d_in[6];
  const float* bv  = (const float*)d_in[7];
  const float* Wo  = (const float*)d_in[8];
  const float* bo  = (const float*)d_in[9];
  const float* Wg  = (const float*)d_in[10];
  const float* bg  = (const float*)d_in[11];
  float* out = (float*)d_out;

  const size_t NE = (size_t)BB * SS * DD;     // 8.4M elements
  _Float16* Qh = (_Float16*)d_ws;
  _Float16* Kh = Qh + NE;
  _Float16* Vt = Kh + NE;
  _Float16* Oh = Vt + NE;
  float* xmean = (float*)(Oh + NE);
  float* gate  = xmean + BB * DD;

  mean_k<<<16, 256, 0, stream>>>(x, xmean);
  gate_k<<<1, 128, 0, stream>>>(xmean, Wg, bg, gate);
  proj_k<float><<<4096, 256, 0, stream>>>(x, Wq, bq, Qh, Vt, out, 0);
  proj_k<float><<<4096, 256, 0, stream>>>(x, Wk, bk, Kh, Vt, out, 0);
  proj_k<float><<<4096, 256, 0, stream>>>(x, Wv, bv, Qh, Vt, out, 1);
  attn_k<<<1024, 128, 0, stream>>>(Qh, Kh, Vt, mid, gate, Oh);
  proj_k<_Float16><<<4096, 256, 0, stream>>>(Oh, Wo, bo, Qh, Vt, out, 2);
}
